// PointsToGaussians_8615704395783
// MI455X (gfx1250) — matrix operand order repaired and checked by reading, not compiled, not run on hardware
//
#include <hip/hip_runtime.h>
#include <math.h>

typedef __attribute__((ext_vector_type(16))) __bf16 v16bf;
typedef __attribute__((ext_vector_type(8)))  float  v8f;

#define ABUF_HALF 8192  // ushorts per 16-row A buffer (16 kt * 32 lanes * 16)

// ---------------------------------------------------------------------------
// Parameter plumbing
// ---------------------------------------------------------------------------
struct HeadParams {
  const uint4 *W1hi, *W1lo, *W2hi, *W2lo, *W3hi, *W3lo; // pre-swizzled bf16 frags
  const float *b1, *g1, *bt1, *b2, *g2, *bt2, *b3;
  int dout;    // real output columns (3,6,1,128)
  int ntile3;  // padded 16-col tiles in stage 3
};
struct KParams {
  const float* feat;
  const float* coord;
  const int*   row_out;   // per point: bid*max_points + pos
  float*       out;
  int          Ntotal;
  long long    S;         // nb * max_points (padded rows per section)
  HeadParams   h[4];      // [0]=offset [1]=covs [2]=opac [3]=feats
};

// ---------------------------------------------------------------------------
// bf16 helpers (bit-exact RNE)
// ---------------------------------------------------------------------------
__device__ inline unsigned short f2bf(float x) {
  union { float f; unsigned u; } v; v.f = x;
  unsigned u = v.u;
  unsigned r = (u + 0x7FFFu + ((u >> 16) & 1u)) >> 16;
  return (unsigned short)r;
}
__device__ inline float bf2f(unsigned short h) {
  union { float f; unsigned u; } v; v.u = ((unsigned)h) << 16; return v.f;
}

__device__ inline v8f wmma_bf16(v16bf a, v16bf b, v8f c) {
  return __builtin_amdgcn_wmma_f32_16x16x32_bf16(false, a, false, b, (short)0, c,
                                                 false, false);
}

// ---------------------------------------------------------------------------
// Shared-LDS A-fragment buffers, pre-swizzled into WMMA lane order.
// Element (m16, k) of a 16xK activation tile lives at:
//   kt = k/32, rem = k%32, hs = (rem>>3)&1, e = (rem&7) + (rem&16 ? 8 : 0),
//   lane = hs*16 + m16, idx = (kt*32 + lane)*16 + e
// so each lane's 16 bf16 values are contiguous (32 B -> 2x ds_load_b128).
// Rows 16..31 use a second buffer at +ABUF_HALF.
// ---------------------------------------------------------------------------
__device__ inline void put_hilo(unsigned short* aHi, unsigned short* aLo, int m,
                                int k, float x) {
  const int kt = k >> 5, rem = k & 31;
  const int hs = (rem >> 3) & 1;
  const int e = (rem & 7) + ((rem & 16) ? 8 : 0);
  const int idx = (m >> 4) * ABUF_HALF + ((kt << 5) + (hs << 4) + (m & 15)) * 16 + e;
  unsigned short h = f2bf(x);
  aHi[idx] = h;
  aLo[idx] = f2bf(x - bf2f(h));
}

__device__ inline v16bf load_afrag(const unsigned short* aBuf, int kt, int lane) {
  union { v16bf v; uint4 q[2]; } u;
  const uint4* p = (const uint4*)(aBuf + ((kt << 5) + lane) * 16);
  u.q[0] = p[0];
  u.q[1] = p[1];
  return u.v;
}

// Load a pre-swizzled 32x16 bf16 B fragment: 32 B/lane, fully coalesced.
__device__ inline v16bf load_bfrag(const uint4* base, long long f, int lane) {
  union { v16bf v; uint4 q[2]; } u;
  const uint4* p = base + f * 64 + (long long)lane * 2;
  u.q[0] = p[0];
  u.q[1] = p[1];
  return u.v;
}

// Dual-M-tile GEMM stage, 3xBF16 split precision. Each B fragment (hi/lo pair)
// feeds 6 WMMAs (2 M-tiles x {Ahi*Bhi, Ahi*Blo, Alo*Bhi}) -> half the L2
// B-traffic per FLOP vs a single 16-row tile.
template <int NT>
__device__ inline void gemm2m(const unsigned short* aHi, const unsigned short* aLo,
                              int numKt, const uint4* Whi, const uint4* Wlo,
                              int ntBase, int lane, v8f* accA, v8f* accB) {
  for (int kt = 0; kt < numKt; ++kt) {
    v16bf a0h = load_afrag(aHi, kt, lane);
    v16bf a0l = load_afrag(aLo, kt, lane);
    v16bf a1h = load_afrag(aHi + ABUF_HALF, kt, lane);
    v16bf a1l = load_afrag(aLo + ABUF_HALF, kt, lane);
#pragma unroll
    for (int t = 0; t < NT; ++t) {
      long long f = (long long)(ntBase + t) * numKt + kt;
      v16bf bhi = load_bfrag(Whi, f, lane);
      v16bf blo = load_bfrag(Wlo, f, lane);
      accA[t] = wmma_bf16(a0h, bhi, accA[t]);
      accA[t] = wmma_bf16(a0h, blo, accA[t]);
      accA[t] = wmma_bf16(a0l, bhi, accA[t]);
      accB[t] = wmma_bf16(a1h, bhi, accB[t]);
      accB[t] = wmma_bf16(a1h, blo, accB[t]);
      accB[t] = wmma_bf16(a1l, bhi, accB[t]);
    }
  }
}

// Accumulator (16x16 f32, VGPR j: lanes0-15 M=j, lanes16-31 M=j+8) -> LDS + bias
__device__ inline void store_tile(float* dst, int stride, const float* bias,
                                  int mloc, int ntile, v8f acc, int lane) {
  const int n = ntile * 16 + (lane & 15), hs = lane >> 4;
  const float b = bias[n];
#pragma unroll
  for (int j = 0; j < 8; ++j) dst[(mloc * 16 + j + 8 * hs) * stride + n] = acc[j] + b;
}

// LayerNorm (eps=1e-5) + exact GELU over 32 x ncols f32 tile in C; result is
// written as pre-swizzled bf16 hi/lo A-fragments (next stage's GEMM input).
// Thread t -> row t>>3 (0..31), segment t&7 (ncols/8 columns each).
__device__ inline void ln_gelu_to_abuf(const float* C, int stride, int ncols,
                                       const float* g, const float* bt, int tid,
                                       unsigned short* aHi, unsigned short* aLo,
                                       float* red_s, float* red_q, float* s_mean,
                                       float* s_rstd) {
  const int r = tid >> 3, seg = tid & 7, per = ncols >> 3;
  const float* row = C + r * stride + seg * per;
  float s = 0.f, q = 0.f;
  for (int i = 0; i < per; ++i) { float x = row[i]; s += x; q += x * x; }
  red_s[tid] = s;
  red_q[tid] = q;
  __syncthreads();
  if (tid < 32) {
    float S = 0.f, Q = 0.f;
    for (int i = 0; i < 8; ++i) { S += red_s[tid * 8 + i]; Q += red_q[tid * 8 + i]; }
    float mean = S / (float)ncols;
    float var  = Q / (float)ncols - mean * mean;
    s_mean[tid] = mean;
    s_rstd[tid] = rsqrtf(var + 1e-5f);
  }
  __syncthreads();
  const float mean = s_mean[r], rs = s_rstd[r];
  const float* gg = g + seg * per;
  const float* bb = bt + seg * per;
  for (int i = 0; i < per; ++i) {
    float y = (row[i] - mean) * rs * gg[i] + bb[i];
    float hgelu = 0.5f * y * (1.f + erff(y * 0.70710678118654752f));
    put_hilo(aHi, aLo, r, seg * per + i, hgelu);
  }
  __syncthreads();
}

__device__ inline float softplusf(float x) { return (x > 20.f) ? x : log1pf(expf(x)); }

// ---------------------------------------------------------------------------
// Weight pre-swizzle: fp32 [K,Nsrc] row-major -> per-lane bf16 hi/lo fragments.
// Fragment f = nt*numKt + kt; element idx = f*512 + lane*16 + e;
// k = kt*32 + (e>>3)*16 + (lane>>4)*8 + (e&7); n = nt*16 + (lane&15). Zero-pad n >= Nsrc.
// ---------------------------------------------------------------------------
__global__ void p2g_swizzle_w(const float* __restrict__ src, int K, int Nsrc, int Npad,
                              unsigned short* __restrict__ hi,
                              unsigned short* __restrict__ lo) {
  const int numKt = K >> 5;
  const long long total = (long long)K * Npad;
  for (long long idx = (long long)blockIdx.x * blockDim.x + threadIdx.x; idx < total;
       idx += (long long)gridDim.x * blockDim.x) {
    int e = (int)(idx & 15);
    int lane = (int)((idx >> 4) & 31);
    long long f = idx >> 9;
    int kt = (int)(f % numKt);
    int nt = (int)(f / numKt);
    int k = kt * 32 + (e >> 3) * 16 + (lane >> 4) * 8 + (e & 7);
    int n = nt * 16 + (lane & 15);
    float v = (n < Nsrc) ? src[(long long)k * Nsrc + n] : 0.f;
    unsigned short h = f2bf(v);
    hi[idx] = h;
    lo[idx] = f2bf(v - bf2f(h));
  }
}

// ---------------------------------------------------------------------------
// Ragged -> padded row map: row_out[pt] = bid*max_points + pos
// ---------------------------------------------------------------------------
__global__ void p2g_rowmap(const int* __restrict__ lengths, int nb,
                           const int* __restrict__ max_points_p, int Ntotal,
                           int* __restrict__ row_out) {
  __shared__ int E[257];
  if (threadIdx.x == 0) {
    int acc = 0;
    for (int i = 0; i < nb; ++i) { E[i] = acc; acc += lengths[i]; }
    E[nb] = acc;
  }
  __syncthreads();
  const int mp = *max_points_p;
  int r = blockIdx.x * blockDim.x + threadIdx.x;
  if (r < Ntotal) {
    int lo = 0, hi = nb;
    while (hi - lo > 1) {
      int mid = (lo + hi) >> 1;
      if (E[mid] <= r) lo = mid; else hi = mid;
    }
    row_out[r] = lo * mp + (r - E[lo]);
  }
}

// ---------------------------------------------------------------------------
// Fused 3-layer MLP (32-point tile, one head) + epilogue + scatter.
// Block = 256 threads = 8 wave32; grid = (mtiles32, 4 heads).
// ---------------------------------------------------------------------------
__global__ __launch_bounds__(256) void p2g_main(KParams kp) {
  __shared__ unsigned short aHi[2 * ABUF_HALF];  // 32 KB (two 16-row tiles)
  __shared__ unsigned short aLo[2 * ABUF_HALF];  // 32 KB
  __shared__ float ldsC[32 * 516];               // f32 accumulator tile (66 KB)
  __shared__ float red_s[256], red_q[256], s_mean[32], s_rstd[32];

  const int tid = threadIdx.x, lane = tid & 31, wave = tid >> 5;
  const int mt = blockIdx.x, head = blockIdx.y;
  const HeadParams H = kp.h[head];
  const int row0 = mt * 32;

  // Load X tile (32 x 256 f32), convert straight into swizzled bf16 hi/lo.
  for (int i = tid; i < 8192; i += 256) {
    int r = i >> 8, k = i & 255, gr = row0 + r;
    float v = (gr < kp.Ntotal) ? kp.feat[(long long)gr * 256 + k] : 0.f;
    put_hilo(aHi, aLo, r, k, v);
  }
  __syncthreads();

  // ---- Stage 1: [32x256] @ [256x512] ----
  v8f accA[4], accB[4];
#pragma unroll
  for (int t = 0; t < 4; ++t) {
    accA[t] = (v8f){0, 0, 0, 0, 0, 0, 0, 0};
    accB[t] = (v8f){0, 0, 0, 0, 0, 0, 0, 0};
  }
  gemm2m<4>(aHi, aLo, 8, H.W1hi, H.W1lo, wave * 4, lane, accA, accB);
#pragma unroll
  for (int t = 0; t < 4; ++t) {
    store_tile(ldsC, 516, H.b1, 0, wave * 4 + t, accA[t], lane);
    store_tile(ldsC, 516, H.b1, 1, wave * 4 + t, accB[t], lane);
  }
  __syncthreads();
  ln_gelu_to_abuf(ldsC, 516, 512, H.g1, H.bt1, tid, aHi, aLo, red_s, red_q, s_mean,
                  s_rstd);

  // ---- Stage 2: [32x512] @ [512x512] ----
#pragma unroll
  for (int t = 0; t < 4; ++t) {
    accA[t] = (v8f){0, 0, 0, 0, 0, 0, 0, 0};
    accB[t] = (v8f){0, 0, 0, 0, 0, 0, 0, 0};
  }
  gemm2m<4>(aHi, aLo, 16, H.W2hi, H.W2lo, wave * 4, lane, accA, accB);
  __syncthreads();
#pragma unroll
  for (int t = 0; t < 4; ++t) {
    store_tile(ldsC, 516, H.b2, 0, wave * 4 + t, accA[t], lane);
    store_tile(ldsC, 516, H.b2, 1, wave * 4 + t, accB[t], lane);
  }
  __syncthreads();
  ln_gelu_to_abuf(ldsC, 516, 512, H.g2, H.bt2, tid, aHi, aLo, red_s, red_q, s_mean,
                  s_rstd);

  // ---- Stage 3: [32x512] @ [512xdout] (wave-uniform predication, EXEC full) ----
  v8f a3A = (v8f){0, 0, 0, 0, 0, 0, 0, 0};
  v8f a3B = (v8f){0, 0, 0, 0, 0, 0, 0, 0};
  if (wave < H.ntile3) {
    gemm2m<1>(aHi, aLo, 16, H.W3hi, H.W3lo, wave, lane, &a3A, &a3B);
    const int n = wave * 16 + (lane & 15), hs = lane >> 4;
    const float b = (n < H.dout) ? H.b3[n] : 0.f;
#pragma unroll
    for (int j = 0; j < 8; ++j) {
      ldsC[(j + 8 * hs) * 128 + n]      = a3A[j] + b;
      ldsC[(16 + j + 8 * hs) * 128 + n] = a3B[j] + b;
    }
  }
  __syncthreads();

  // ---- Head epilogue + scatter into padded output ----
  float* out = kp.out;
  const long long S = kp.S;
  if (head == 0) {  // offset -> means (sec0), offs (sec1)
    if (tid < 32) {
      int gr = row0 + tid;
      if (gr < kp.Ntotal) {
        long long ro = kp.row_out[gr];
#pragma unroll
        for (int c = 0; c < 3; ++c) {
          float o = ldsC[tid * 128 + c];
          out[ro * 3 + c] = kp.coord[(long long)gr * 3 + c] + o;
          out[S * 3 + ro * 3 + c] = o;
        }
      }
    }
  } else if (head == 1) {  // covs -> sec3
    if (tid < 32) {
      int gr = row0 + tid;
      if (gr < kp.Ntotal) {
        long long ro = kp.row_out[gr];
        float c0 = ldsC[tid * 128 + 0], c1 = ldsC[tid * 128 + 1];
        float c2 = ldsC[tid * 128 + 2], c3 = ldsC[tid * 128 + 3];
        float c4 = ldsC[tid * 128 + 4], c5 = ldsC[tid * 128 + 5];
        float xx = softplusf(c0) + 1e-4f;
        float yy = softplusf(c3) + 1e-4f;
        float zz = softplusf(c5) + 1e-4f;
        float xy = tanhf(c1) * sqrtf(xx * yy) * 0.9f;
        float xz = tanhf(c2) * sqrtf(xx * zz) * 0.9f;
        float yz = tanhf(c4) * sqrtf(yy * zz) * 0.9f;
        float* o = out + S * 134 + ro * 6;
        o[0] = xx; o[1] = xy; o[2] = xz; o[3] = yy; o[4] = yz; o[5] = zz;
      }
    }
  } else if (head == 2) {  // opac -> sec4
    if (tid < 32) {
      int gr = row0 + tid;
      if (gr < kp.Ntotal) {
        long long ro = kp.row_out[gr];
        float x = ldsC[tid * 128];
        out[S * 140 + ro] = 1.f / (1.f + expf(-x));
      }
    }
  } else {  // feats (128 cols) -> sec2
    int r = tid >> 3, gr = row0 + r;
    if (gr < kp.Ntotal) {
      long long ro = kp.row_out[gr];
      int c0 = (tid & 7) * 16;
#pragma unroll
      for (int i = 0; i < 16; ++i)
        out[S * 6 + ro * 128 + c0 + i] = ldsC[r * 128 + c0 + i];
    }
  }
}

// ---------------------------------------------------------------------------
// Host launcher
// d_in flatten order: feat, coord, lengths, max_points, then params
// (feats, offset, covs, opac) x (W1,b1,g1,bt1,W2,b2,g2,bt2,W3,b3).
// ---------------------------------------------------------------------------
extern "C" void kernel_launch(void* const* d_in, const int* in_sizes, int n_in,
                              void* d_out, int out_size, void* d_ws, size_t ws_size,
                              hipStream_t stream) {
  const float* feat   = (const float*)d_in[0];
  const float* coord  = (const float*)d_in[1];
  const int*  lengths = (const int*)d_in[2];
  const int*  maxp    = (const int*)d_in[3];
  const int nb = in_sizes[2];
  const int Ntotal = in_sizes[0] / 256;
  const long long S = out_size / 141;  // nb * max_points

  hipMemsetAsync(d_out, 0, (size_t)out_size * sizeof(float), stream);
  if (Ntotal <= 0) return;

  char* ws = (char*)d_ws;
  size_t cur = 0;
  auto alloc = [&](size_t bytes) -> char* {
    char* p = ws + cur;
    cur = (cur + bytes + 255) & ~(size_t)255;
    return p;
  };

  int* row_out = (int*)alloc((size_t)Ntotal * sizeof(int));
  p2g_rowmap<<<(Ntotal + 255) / 256, 256, 0, stream>>>(lengths, nb, maxp, Ntotal,
                                                       row_out);

  KParams kp;
  kp.feat = feat;
  kp.coord = coord;
  kp.row_out = row_out;
  kp.out = (float*)d_out;
  kp.Ntotal = Ntotal;
  kp.S = S;

  // Kernel head order: 0=offset, 1=covs, 2=opac, 3=feats
  const int bases[4] = {14, 24, 34, 4};
  for (int i = 0; i < 4; ++i) {
    const int b0 = bases[i];
    const float* W1 = (const float*)d_in[b0 + 0];
    const float* W2 = (const float*)d_in[b0 + 4];
    const float* W3 = (const float*)d_in[b0 + 8];
    const int dout = in_sizes[b0 + 9];
    const int npad = (dout + 15) & ~15;

    HeadParams& H = kp.h[i];
    H.b1 = (const float*)d_in[b0 + 1];
    H.g1 = (const float*)d_in[b0 + 2];
    H.bt1 = (const float*)d_in[b0 + 3];
    H.b2 = (const float*)d_in[b0 + 5];
    H.g2 = (const float*)d_in[b0 + 6];
    H.bt2 = (const float*)d_in[b0 + 7];
    H.b3 = (const float*)d_in[b0 + 9];
    H.dout = dout;
    H.ntile3 = npad / 16;

    const size_t e1 = (size_t)256 * 512;
    const size_t e2 = (size_t)512 * 512;
    const size_t e3 = (size_t)512 * npad;
    unsigned short* w1h = (unsigned short*)alloc(e1 * 2);
    unsigned short* w1l = (unsigned short*)alloc(e1 * 2);
    unsigned short* w2h = (unsigned short*)alloc(e2 * 2);
    unsigned short* w2l = (unsigned short*)alloc(e2 * 2);
    unsigned short* w3h = (unsigned short*)alloc(e3 * 2);
    unsigned short* w3l = (unsigned short*)alloc(e3 * 2);

    p2g_swizzle_w<<<(int)((e1 + 255) / 256), 256, 0, stream>>>(W1, 256, 512, 512, w1h, w1l);
    p2g_swizzle_w<<<(int)((e2 + 255) / 256), 256, 0, stream>>>(W2, 512, 512, 512, w2h, w2l);
    p2g_swizzle_w<<<(int)((e3 + 255) / 256), 256, 0, stream>>>(W3, 512, dout, npad, w3h, w3l);

    H.W1hi = (const uint4*)w1h;
    H.W1lo = (const uint4*)w1l;
    H.W2hi = (const uint4*)w2h;
    H.W2lo = (const uint4*)w2l;
    H.W3hi = (const uint4*)w3h;
    H.W3lo = (const uint4*)w3l;
  }

  dim3 grid((Ntotal + 31) / 32, 4);
  p2g_main<<<grid, 256, 0, stream>>>(kp);
}
